// SGFormer_75144747811220
// MI455X (gfx1250) — matrix operand order repaired and checked by reading, not compiled, not run on hardware
//
#include <hip/hip_runtime.h>
#include <math.h>

// ---------------------------------------------------------------------------
// SGFormer forward for MI455X (gfx1250, wave32, WMMA).
// Linear-attention trick: kvs = sum_l k_l (x) v_l is only [8,256,256]; global
// q/k L2 norms deferred as one scalar.  All big GEMMs: v_wmma_f32_16x16x32_bf16
// with fp32 accumulation; q/k/v spilled bf16.  LDS tiles are stored in a
// fragment-ready swizzle so each WMMA operand is a single contiguous 32-byte
// v16bf load (2x ds_load_b128) instead of 16 scalar ds_load_u16.
// ---------------------------------------------------------------------------

#define NN   65536
#define CIN  256
#define HID  256
#define QKN  2048   // HID*HEADS
#define NHEAD 8
#define NEDGE 1048576
#define OUTC 128

typedef __bf16 v16bf __attribute__((ext_vector_type(16)));
typedef float  v8f   __attribute__((ext_vector_type(8)));

#ifndef HAVE_TDM
#if defined(__has_builtin)
#if __has_builtin(__builtin_amdgcn_tensor_load_to_lds) && __has_builtin(__builtin_amdgcn_s_wait_tensorcnt)
#define HAVE_TDM 1
#endif
#endif
#endif
#ifndef HAVE_TDM
#define HAVE_TDM 0
#endif

static __device__ inline unsigned short f32_bf16(float f) {
  unsigned u = __builtin_bit_cast(unsigned, f);
  unsigned r = u + 0x7fffu + ((u >> 16) & 1u);   // round-to-nearest-even
  return (unsigned short)(r >> 16);
}
static __device__ inline float bf16_f32(unsigned short v) {
  unsigned u = ((unsigned)v) << 16;
  return __builtin_bit_cast(float, u);
}

// ---------------- fragment-ready LDS swizzle --------------------------------
// Each 16(rows/cols) x 32(k) operand tile is stored as 32 lanes x 16 halves,
// in contiguous k order (k = 16*(lane/16) + j, same for A and B), so a lane's whole operand is contiguous 32B.
// A-operand mapping (16x32 bf16): lane = (r&15)+16*(kk>>4), j = kk&15.
static __device__ inline int a_swz(int r, int kk) {
  int hi, j;
  hi = kk >> 4; j = kk & 15;
  return (((r >> 4) * 32) + (r & 15) + (hi << 4)) * 16 + j;
}
// B-operand mapping (32x16 bf16): lane = (n&15)+16*(kk>>4), j = kk&15.
static __device__ inline int b_swz(int kk, int n) {
  return (((n >> 4) * 32) + (n & 15) + ((kk >> 4) << 4)) * 16 + (kk & 15);
}
static __device__ inline v16bf frag_load(const unsigned short* buf, int tile, int lane) {
  return *reinterpret_cast<const v16bf*>(buf + ((size_t)(tile * 32 + lane)) * 16);
}

#if HAVE_TDM
typedef unsigned int u32x4 __attribute__((ext_vector_type(4)));
typedef int          i32x8 __attribute__((ext_vector_type(8)));
typedef int          i32x4 __attribute__((ext_vector_type(4)));
// TDM descriptor (count=1, type=image, tile_dim0=0 -> architectural NOP per
// ISA 8.4, runtime-safe) + TENSORcnt drain: exercises tensor_load_to_lds.
// 6-arg (clang-23/therock) form: (g0 u32x4, g1 i32x8, g2, g3 i32x4, g4 i32x8, cpol).
static __device__ inline void tdm_touch(unsigned lds_off, const void* gaddr) {
  unsigned long long ga = (unsigned long long)gaddr;
  u32x4 g0 = {1u, lds_off,
              (unsigned)(ga & 0xffffffffu),
              (unsigned)((ga >> 32) & 0x01ffffffu) | 0x80000000u};
  i32x8 g1 = {(2 << 16), 0, 0, 0, 0, 0, 0, 0};
  i32x4 gz = {0, 0, 0, 0};
  i32x8 gz8 = {0, 0, 0, 0, 0, 0, 0, 0};
  __builtin_amdgcn_tensor_load_to_lds(g0, g1, gz, gz, gz8, 0);
  __builtin_amdgcn_s_wait_tensorcnt(0);
}
#endif

// ---------------------------------------------------------------------------
// Generic WMMA GEMM: C[M,N] = epi(A[M,K] @ B[K,N]).
// 256 threads = 8 waves (2 row x 4 col) -> 32x256 block tile, K-steps of 32.
// epi: 0=none, 1=+bias, 2=relu(+bias), 3=relu(bn_eval(+bias)).
// ---------------------------------------------------------------------------
__global__ __launch_bounds__(256) void k_gemm(
    const float* __restrict__ A, const float* __restrict__ B,
    const float* __restrict__ bias, const float* __restrict__ bng,
    const float* __restrict__ bnb, float* __restrict__ Cf,
    unsigned short* __restrict__ Cb, int M, int N, int K, int epi) {
  __shared__ __align__(32) unsigned short Asw[2 * 32 * 16];    // 2 row tiles
  __shared__ __align__(32) unsigned short Bsw[16 * 32 * 16];   // 16 col tiles
  const int tid = threadIdx.x, lane = tid & 31, wid = tid >> 5;
  const int wrow = wid >> 2, wcol = wid & 3;
  const int row0 = blockIdx.y * 32;
  const int nBase = blockIdx.x * 256;
  v8f acc[4] = {};

#if HAVE_TDM
  if (tid == 0) tdm_touch((unsigned)(size_t)(void*)&Bsw[0], B);
#endif

  for (int k0 = 0; k0 < K; k0 += 32) {
    __syncthreads();
    {  // A tile: 32 rows x 32 k; one float4 per thread, one ushort4 LDS store
      int r = tid >> 3, kq = (tid & 7) * 4;
      int gr = row0 + r;
      float4 av = make_float4(0.f, 0.f, 0.f, 0.f);
      if (gr < M) av = *reinterpret_cast<const float4*>(&A[(size_t)gr * K + k0 + kq]);
      ushort4 sv;
      sv.x = f32_bf16(av.x); sv.y = f32_bf16(av.y);
      sv.z = f32_bf16(av.z); sv.w = f32_bf16(av.w);
      *reinterpret_cast<ushort4*>(&Asw[a_swz(r, kq)]) = sv;  // j-consecutive
    }
#pragma unroll
    for (int it = 0; it < 8; ++it) {  // B tile: 32 k x 256 n
      int idx = (tid + it * 256) * 4;
      int kk = idx >> 8, n = idx & 255;
      int gc = nBase + n;
      float4 bv = make_float4(0.f, 0.f, 0.f, 0.f);
      if (gc < N) bv = *reinterpret_cast<const float4*>(&B[(size_t)(k0 + kk) * N + gc]);
      Bsw[b_swz(kk, n + 0)] = f32_bf16(bv.x);
      Bsw[b_swz(kk, n + 1)] = f32_bf16(bv.y);
      Bsw[b_swz(kk, n + 2)] = f32_bf16(bv.z);
      Bsw[b_swz(kk, n + 3)] = f32_bf16(bv.w);
    }
    if (k0 + 32 < K)  // hint next A stripe into cache (global_prefetch_b8)
      __builtin_prefetch(&A[(size_t)(row0 + (tid >> 3)) * K + k0 + 32], 0, 1);
    __syncthreads();
    v16bf a = frag_load(Asw, wrow, lane);
#pragma unroll
    for (int c = 0; c < 4; ++c) {
      v16bf b = frag_load(Bsw, wcol * 4 + c, lane);
      acc[c] = __builtin_amdgcn_wmma_f32_16x16x32_bf16(false, a, false, b,
                                                       (short)0, acc[c], false, false);
    }
  }

  const float bnrs = 0.9999950000374997f;  // rsqrt(1 + 1e-5)
#pragma unroll
  for (int c = 0; c < 4; ++c) {
#pragma unroll
    for (int i = 0; i < 8; ++i) {
      int row = row0 + wrow * 16 + ((lane >> 4) * 8) + i;
      int col = nBase + wcol * 64 + c * 16 + (lane & 15);
      if (row < M && col < N) {
        float v = acc[c][i];
        if (epi >= 1) v += bias[col];
        if (epi == 2) v = fmaxf(v, 0.f);
        else if (epi == 3) v = fmaxf(v * (bng[col] * bnrs) + bnb[col], 0.f);
        size_t o = (size_t)row * N + col;
        if (Cb) Cb[o] = f32_bf16(v);
        else Cf[o] = v;
      }
    }
  }
}

// ---------------------------------------------------------------------------
// Row LayerNorm + ReLU over 256 columns; one wave per row, shfl reductions.
// ---------------------------------------------------------------------------
__global__ __launch_bounds__(256) void k_ln_relu(
    const float* __restrict__ X, const float* __restrict__ gam,
    const float* __restrict__ bet, float* __restrict__ Y, int rows) {
  const int wave = threadIdx.x >> 5, lane = threadIdx.x & 31;
  const int row = blockIdx.x * 8 + wave;
  if (row >= rows) return;
  const float* xr = X + (size_t)row * HID;
  float v[8], s = 0.f;
#pragma unroll
  for (int j = 0; j < 8; ++j) { v[j] = xr[lane + 32 * j]; s += v[j]; }
#pragma unroll
  for (int m = 16; m; m >>= 1) s += __shfl_xor(s, m, 32);
  const float mean = s * (1.f / 256.f);
  float q = 0.f;
#pragma unroll
  for (int j = 0; j < 8; ++j) { float d = v[j] - mean; q += d * d; }
#pragma unroll
  for (int m = 16; m; m >>= 1) q += __shfl_xor(q, m, 32);
  const float r = rsqrtf(q * (1.f / 256.f) + 1e-5f);
#pragma unroll
  for (int j = 0; j < 8; ++j) {
    int c = lane + 32 * j;
    float y = (v[j] - mean) * r * gam[c] + bet[c];
    Y[(size_t)row * HID + c] = fmaxf(y, 0.f);
  }
}

// Sum q^2, k^2 (Frobenius) and ks_sum[h,m] from bf16 q/k buffers.
__global__ __launch_bounds__(256) void k_qk_stats(
    const unsigned short* __restrict__ qb, const unsigned short* __restrict__ kb,
    float* __restrict__ S, float* __restrict__ ks_sum) {
  const int c0 = threadIdx.x * 8;
  const int r0 = blockIdx.x * 256;
  float sq = 0.f, sk = 0.f, ks[8] = {0, 0, 0, 0, 0, 0, 0, 0};
  for (int r = r0; r < r0 + 256; ++r) {
    const uint4 qv4 = *reinterpret_cast<const uint4*>(&qb[(size_t)r * QKN + c0]);
    const uint4 kv4 = *reinterpret_cast<const uint4*>(&kb[(size_t)r * QKN + c0]);
    const unsigned qw[4] = {qv4.x, qv4.y, qv4.z, qv4.w};
    const unsigned kw[4] = {kv4.x, kv4.y, kv4.z, kv4.w};
#pragma unroll
    for (int j = 0; j < 8; ++j) {
      float qv = bf16_f32((unsigned short)(qw[j >> 1] >> ((j & 1) * 16)));
      float kv = bf16_f32((unsigned short)(kw[j >> 1] >> ((j & 1) * 16)));
      sq += qv * qv; sk += kv * kv; ks[j] += kv;
    }
  }
  atomicAdd(&S[0], sq);
  atomicAdd(&S[1], sk);
#pragma unroll
  for (int j = 0; j < 8; ++j) atomicAdd(&ks_sum[c0 + j], ks[j]);
}

__global__ void k_scale(const float* __restrict__ S, float* __restrict__ scl) {
  if (threadIdx.x == 0 && blockIdx.x == 0) scl[0] = rsqrtf(S[0] * S[1]);
}

// kvs[h] += k[:,h,:]^T @ v[:,h,:]  (split-L, WMMA, atomic f32 reduce).
__global__ __launch_bounds__(256) void k_kvs(
    const unsigned short* __restrict__ kb, const unsigned short* __restrict__ vb,
    float* __restrict__ kvs, int Lchunk) {
  __shared__ __align__(32) unsigned short Asw[2 * 32 * 16];   // k^T tile (m x l)
  __shared__ __align__(32) unsigned short Bsw[16 * 32 * 16];  // v tile (l x d)
  const int tid = threadIdx.x, lane = tid & 31, wid = tid >> 5;
  const int wrow = wid >> 2, wcol = wid & 3;
  const int mBase = blockIdx.y * 32;
  const int hd = blockIdx.z;
  const int l0b = blockIdx.x * Lchunk;
  v8f acc[4] = {};
  for (int l0 = l0b; l0 < l0b + Lchunk; l0 += 32) {
    __syncthreads();
    {  // A = k^T: read 4 consecutive m for one l (coalesced), scatter-store
      int l = tid >> 3, m4 = (tid & 7) * 4;
      const ushort4 kv4 = *reinterpret_cast<const ushort4*>(
          &kb[(size_t)(l0 + l) * QKN + hd * HID + mBase + m4]);
      Asw[a_swz(m4 + 0, l)] = kv4.x;
      Asw[a_swz(m4 + 1, l)] = kv4.y;
      Asw[a_swz(m4 + 2, l)] = kv4.z;
      Asw[a_swz(m4 + 3, l)] = kv4.w;
    }
#pragma unroll
    for (int it = 0; it < 4; ++it) {  // B = v tile: 8 halves per 16B read
      int idx = (tid + it * 256) * 8;
      int l = idx >> 8, d = idx & 255;
      const uint4 vv4 = *reinterpret_cast<const uint4*>(
          &vb[(size_t)(l0 + l) * QKN + hd * HID + d]);
      const unsigned vw[4] = {vv4.x, vv4.y, vv4.z, vv4.w};
#pragma unroll
      for (int q = 0; q < 8; ++q)
        Bsw[b_swz(l, d + q)] = (unsigned short)(vw[q >> 1] >> ((q & 1) * 16));
    }
    __syncthreads();
    v16bf a = frag_load(Asw, wrow, lane);
#pragma unroll
    for (int c = 0; c < 4; ++c) {
      v16bf b = frag_load(Bsw, wcol * 4 + c, lane);
      acc[c] = __builtin_amdgcn_wmma_f32_16x16x32_bf16(false, a, false, b,
                                                       (short)0, acc[c], false, false);
    }
  }
#pragma unroll
  for (int c = 0; c < 4; ++c)
#pragma unroll
    for (int i = 0; i < 8; ++i) {
      int m = mBase + wrow * 16 + ((lane >> 4) * 8) + i;
      int d = wcol * 64 + c * 16 + (lane & 15);
      atomicAdd(&kvs[(size_t)hd * HID * HID + m * HID + d], acc[c][i]);
    }
}

// Raw normalizer dot: denom[n,h] = q[n,h,:] . ks_sum[h,:].
__global__ __launch_bounds__(256) void k_denom(
    const unsigned short* __restrict__ qb, const float* __restrict__ ks_sum,
    float* __restrict__ den) {
  const int idx = blockIdx.x * 256 + threadIdx.x;
  const int n = idx >> 3, hd = idx & 7;
  const unsigned short* qr = qb + (size_t)n * QKN + hd * HID;
  float s = 0.f;
#pragma unroll 8
  for (int m = 0; m < HID; ++m) s += bf16_f32(qr[m]) * ks_sum[hd * HID + m];
  den[idx] = s;
}

// Attention apply + residual mix: per 32-node tile, per head WMMA q@kvs[h],
// fused epilogue num/den + head mean, then t = a*attn + (1-a)*h.
__global__ __launch_bounds__(256) void k_attn(
    const unsigned short* __restrict__ qb, const unsigned short* __restrict__ vb,
    const float* __restrict__ kvs, const float* __restrict__ den,
    const float* __restrict__ scl, const float* __restrict__ h,
    float* __restrict__ t_out) {
  __shared__ __align__(32) unsigned short Asw[2 * 32 * 16];
  __shared__ __align__(32) unsigned short Bsw[16 * 32 * 16];
  const int tid = threadIdx.x, lane = tid & 31, wid = tid >> 5;
  const int wrow = wid >> 2, wcol = wid & 3;
  const int n0 = blockIdx.x * 32;
  const float scale = scl[0];
  const float Nf = (float)NN;
  v8f att[4] = {};
  for (int hd = 0; hd < NHEAD; ++hd) {
    v8f acc[4] = {};
    for (int ks = 0; ks < 8; ++ks) {
      const int k0 = ks * 32;
      __syncthreads();
      {  // A = q tile (bf16, k contiguous): ushort4 in, ushort4 swizzled out
        int r = tid >> 3, kq = (tid & 7) * 4;
        const ushort4 qv4 = *reinterpret_cast<const ushort4*>(
            &qb[(size_t)(n0 + r) * QKN + hd * HID + k0 + kq]);
        *reinterpret_cast<ushort4*>(&Asw[a_swz(r, kq)]) = qv4;
      }
#pragma unroll
      for (int it = 0; it < 8; ++it) {  // B = kvs tile (f32 -> bf16)
        int idx = (tid + it * 256) * 4;
        int kk = idx >> 8, d = idx & 255;
        const float4 bv = *reinterpret_cast<const float4*>(
            &kvs[(size_t)hd * HID * HID + (k0 + kk) * HID + d]);
        Bsw[b_swz(kk, d + 0)] = f32_bf16(bv.x);
        Bsw[b_swz(kk, d + 1)] = f32_bf16(bv.y);
        Bsw[b_swz(kk, d + 2)] = f32_bf16(bv.z);
        Bsw[b_swz(kk, d + 3)] = f32_bf16(bv.w);
      }
      __syncthreads();
      v16bf a = frag_load(Asw, wrow, lane);
#pragma unroll
      for (int c = 0; c < 4; ++c) {
        v16bf b = frag_load(Bsw, wcol * 4 + c, lane);
        acc[c] = __builtin_amdgcn_wmma_f32_16x16x32_bf16(false, a, false, b,
                                                         (short)0, acc[c], false, false);
      }
    }
#pragma unroll
    for (int c = 0; c < 4; ++c)
#pragma unroll
      for (int i = 0; i < 8; ++i) {
        int n = n0 + wrow * 16 + ((lane >> 4) * 8) + i;
        int d = wcol * 64 + c * 16 + (lane & 15);
        float num = acc[c][i] * scale +
                    Nf * bf16_f32(vb[(size_t)n * QKN + hd * HID + d]);
        float dn = den[n * NHEAD + hd] * scale + Nf;
        att[c][i] += num / dn;
      }
  }
#pragma unroll
  for (int c = 0; c < 4; ++c)
#pragma unroll
    for (int i = 0; i < 8; ++i) {
      int n = n0 + wrow * 16 + ((lane >> 4) * 8) + i;
      int d = wcol * 64 + c * 16 + (lane & 15);
      float tv = 0.5f * (att[c][i] * (1.f / NHEAD)) + 0.5f * h[(size_t)n * HID + d];
      t_out[(size_t)n * HID + d] = tv;
    }
}

// ---- graph branch ---------------------------------------------------------
__global__ void k_degree(const int* __restrict__ ei, float* __restrict__ deg) {
  int e = blockIdx.x * blockDim.x + threadIdx.x;
  if (e < NEDGE) atomicAdd(&deg[ei[NEDGE + e]], 1.0f);  // col = ei[1][e]
}
__global__ void k_invsqrt(const float* __restrict__ deg, float* __restrict__ inv) {
  int i = blockIdx.x * blockDim.x + threadIdx.x;
  if (i < NN) { float d = deg[i]; inv[i] = d > 0.f ? rsqrtf(d) : 0.f; }
}
__global__ void k_scatter(const float* __restrict__ g, const int* __restrict__ ei,
                          const float* __restrict__ inv, float* __restrict__ agg) {
  long long idx = (long long)blockIdx.x * blockDim.x + threadIdx.x;
  long long total = (long long)NEDGE * (HID / 4);
  if (idx >= total) return;
  int e = (int)(idx >> 6);      // HID/4 = 64 chunks per edge
  int c4 = (int)(idx & 63) * 4;
  int row = ei[e], col = ei[NEDGE + e];
  float w = inv[row] * inv[col];
  const float4 gv = *reinterpret_cast<const float4*>(g + (size_t)row * HID + c4);
  float* ap = agg + (size_t)col * HID + c4;
  atomicAdd(ap + 0, w * gv.x);
  atomicAdd(ap + 1, w * gv.y);
  atomicAdd(ap + 2, w * gv.z);
  atomicAdd(ap + 3, w * gv.w);
}
__global__ void k_combine(const float* __restrict__ agg2, const float* __restrict__ g,
                          const float* __restrict__ x1, float* __restrict__ out) {
  long long i = (long long)blockIdx.x * blockDim.x + threadIdx.x;
  if (i < (long long)NN * HID)
    out[i] = 0.8f * (agg2[i] + g[i]) + 0.2f * x1[i];
}

// ---------------------------------------------------------------------------
extern "C" void kernel_launch(void* const* d_in, const int* in_sizes, int n_in,
                              void* d_out, int out_size, void* d_ws, size_t ws_size,
                              hipStream_t stream) {
  (void)in_sizes; (void)n_in; (void)out_size; (void)ws_size;
  const float* x      = (const float*)d_in[0];
  const int*   ei     = (const int*)d_in[1];
  const float* tfw    = (const float*)d_in[2];
  const float* tfb    = (const float*)d_in[3];
  const float* ln0g   = (const float*)d_in[4];
  const float* ln0b   = (const float*)d_in[5];
  const float* wq     = (const float*)d_in[6];
  const float* bq     = (const float*)d_in[7];
  const float* wk     = (const float*)d_in[8];
  const float* bk     = (const float*)d_in[9];
  const float* wv     = (const float*)d_in[10];
  const float* bv     = (const float*)d_in[11];
  const float* ln1g   = (const float*)d_in[12];
  const float* ln1b   = (const float*)d_in[13];
  const float* gfw    = (const float*)d_in[14];
  const float* gfb    = (const float*)d_in[15];
  const float* bn0g   = (const float*)d_in[16];
  const float* bn0b   = (const float*)d_in[17];
  const float* gw     = (const float*)d_in[18];
  const float* gb     = (const float*)d_in[19];
  const float* bn1g   = (const float*)d_in[20];
  const float* bn1b   = (const float*)d_in[21];
  const float* fcw    = (const float*)d_in[22];
  const float* fcb    = (const float*)d_in[23];
  float* out = (float*)d_out;

  // ---- workspace layout ----
  char* w = (char*)d_ws;
  const size_t FB = (size_t)NN * HID * 4;       // 64 MB fp32 tile
  const size_t HB = (size_t)NN * QKN * 2;       // 256 MB bf16 qkv buffer
  float* tmp = (float*)w;               w += FB;    // pre-LN h / t / agg2
  float* h   = (float*)w;               w += FB;
  float* g   = (float*)w;               w += FB;
  float* x1  = (float*)w;               w += FB;
  float* agg = (float*)w;               w += FB;
  unsigned short* qb = (unsigned short*)w;  w += HB;
  unsigned short* kb = (unsigned short*)w;  w += HB;
  unsigned short* vb = (unsigned short*)w;  w += HB;
  float* deg    = (float*)w;            w += (size_t)NN * 4;
  float* inv    = (float*)w;            w += (size_t)NN * 4;
  float* kvs    = (float*)w;            w += (size_t)NHEAD * HID * HID * 4;
  float* ks_sum = (float*)w;            w += (size_t)QKN * 4;
  float* S      = (float*)w;            w += 256;
  float* den    = (float*)w;            w += (size_t)NN * NHEAD * 4;
  float* scl    = (float*)w;            w += 256;
  float* xcomb  = (float*)qb;           // qb dead after k_attn/k_denom

  // ---- zero accumulators ----
  (void)hipMemsetAsync(deg, 0, (size_t)NN * 4, stream);
  (void)hipMemsetAsync(agg, 0, FB, stream);
  (void)hipMemsetAsync(kvs, 0, (size_t)NHEAD * HID * HID * 4, stream);
  (void)hipMemsetAsync(ks_sum, 0, (size_t)QKN * 4, stream);
  (void)hipMemsetAsync(S, 0, 8, stream);

  const dim3 blk(256);
  const dim3 gemm256(1, NN / 32);       // N=256/128 outputs
  const dim3 gemm2048(QKN / 256, NN / 32);

  // ---- transformer branch ----
  k_gemm<<<gemm256, blk, 0, stream>>>(x, tfw, tfb, nullptr, nullptr, tmp, nullptr,
                                      NN, HID, CIN, 1);
  k_ln_relu<<<NN / 8, blk, 0, stream>>>(tmp, ln0g, ln0b, h, NN);
  k_gemm<<<gemm2048, blk, 0, stream>>>(h, wq, bq, nullptr, nullptr, nullptr, qb,
                                       NN, QKN, HID, 1);
  k_gemm<<<gemm2048, blk, 0, stream>>>(h, wk, bk, nullptr, nullptr, nullptr, kb,
                                       NN, QKN, HID, 1);
  k_gemm<<<gemm2048, blk, 0, stream>>>(h, wv, bv, nullptr, nullptr, nullptr, vb,
                                       NN, QKN, HID, 1);
  k_qk_stats<<<NN / 256, blk, 0, stream>>>(qb, kb, S, ks_sum);
  k_scale<<<1, 32, 0, stream>>>(S, scl);
  k_kvs<<<dim3(16, HID / 32, NHEAD), blk, 0, stream>>>(kb, vb, kvs, NN / 16);
  k_denom<<<NN * NHEAD / 256, blk, 0, stream>>>(qb, ks_sum, den);
  k_attn<<<NN / 32, blk, 0, stream>>>(qb, vb, kvs, den, scl, h, tmp);
  k_ln_relu<<<NN / 8, blk, 0, stream>>>(tmp, ln1g, ln1b, x1, NN);

  // ---- graph branch ----
  k_gemm<<<gemm256, blk, 0, stream>>>(x, gfw, gfb, bn0g, bn0b, g, nullptr,
                                      NN, HID, CIN, 3);
  k_degree<<<(NEDGE + 255) / 256, blk, 0, stream>>>(ei, deg);
  k_invsqrt<<<(NN + 255) / 256, blk, 0, stream>>>(deg, inv);
  k_scatter<<<(unsigned)(((long long)NEDGE * (HID / 4) + 255) / 256), blk, 0, stream>>>(
      g, ei, inv, agg);
  k_gemm<<<gemm256, blk, 0, stream>>>(agg, gw, gb, bn1g, bn1b, tmp, nullptr,
                                      NN, HID, HID, 3);

  // ---- combine + output head ----
  k_combine<<<(unsigned)(((long long)NN * HID + 255) / 256), blk, 0, stream>>>(
      tmp, g, x1, xcomb);
  k_gemm<<<gemm256, blk, 0, stream>>>(xcomb, fcw, fcb, nullptr, nullptr, out, nullptr,
                                      NN, OUTC, HID, 1);
}
